// StandardCSSM_49125835931747
// MI455X (gfx1250) — matrix operand order repaired and checked by reading, not compiled, not run on hardware
//
#include <hip/hip_runtime.h>
#include <math.h>

// ---------------------------------------------------------------------------
// StandardCSSM on MI455X (gfx1250):
//   rfft2 -> per-bin complex scan over T -> irfft2
// FFTs are dense 64-pt DFT matmuls on v_wmma_f32_16x16x32_f16 using split-f16
// (hi+lo) operands with f32 accumulation (~f32 accuracy, 3 WMMAs per product).
// Bulk LDS fills use GLOBAL_LOAD_ASYNC_TO_LDS (ASYNCcnt) on gfx1250.
// The time scan is a direct complex recurrence (mathematically equal to the
// reference's log-domain associative scan).
// ---------------------------------------------------------------------------

typedef _Float16 half_t;
typedef __attribute__((ext_vector_type(16))) _Float16 v16h;
typedef __attribute__((ext_vector_type(8)))  _Float16 v8h;
typedef __attribute__((ext_vector_type(8)))  float    v8f;

#define TDIM 64
#define BDIM 2
#define CDIM 64
#define HDIM 64
#define WDIM 64
#define WFD  33   // rfft width
#define WFP  48   // padded to 3 tiles of 16

static constexpr size_t NBINS   = (size_t)BDIM * CDIM * HDIM * WFD;   // 270336
static constexpr size_t U_BYTES = NBINS * TDIM * sizeof(float2);      // 138412032
static constexpr size_t K_ELEMS = (size_t)CDIM * HDIM * WFD;          // 135168
static constexpr size_t K_BYTES = K_ELEMS * sizeof(float2);

#define VZERO8 {0.f,0.f,0.f,0.f,0.f,0.f,0.f,0.f}

__device__ __forceinline__ void splitf(float v, half_t& hi, half_t& lo) {
  half_t h = (half_t)v;
  hi = h;
  lo = (half_t)(v - (float)h);
}

__device__ __forceinline__ v8f wmma16(v16h a, v16h b, v8f c) {
  // 8 args: (neg_a, A, neg_b, B, c_mod, C, reuse_a, reuse_b); NEG must be 0 for f16
  return __builtin_amdgcn_wmma_f32_16x16x32_f16(false, a, false, b, (short)0, c,
                                                false, false);
}

// split-f16 product: (ah+al)*(bh+bl) ~= ah*bh + ah*bl + al*bh  (f32 accumulate)
__device__ __forceinline__ v8f mma_split(v16h ah, v16h al, v16h bh, v16h bl, v8f acc) {
  acc = wmma16(ah, bh, acc);
  acc = wmma16(ah, bl, acc);
  acc = wmma16(al, bh, acc);
  return acc;
}

// A fragment: row-major source, 16x32 f16 tile at (mr, kb), leading dim ld.
__device__ __forceinline__ v16h load_a16(const half_t* base, int ld, int mr, int kb, int lane) {
  int row = mr + (lane & 15);
  int ko  = kb + ((lane >> 4) << 3);
  const half_t* p = base + row * ld + ko;
  v16h r;
  ((v8h*)&r)[0] = *(const v8h*)p;
  ((v8h*)&r)[1] = *(const v8h*)(p + 16);
  return r;
}

// B fragment: source stored TRANSPOSED (row-major R^T, ld over K).
__device__ __forceinline__ v16h load_b16(const half_t* baseT, int ld, int kb, int nb, int lane) {
  int col = nb + (lane & 15);
  int ko  = kb + ((lane >> 4) << 3);
  const half_t* p = baseT + col * ld + ko;
  v16h r;
  ((v8h*)&r)[0] = *(const v8h*)p;
  ((v8h*)&r)[1] = *(const v8h*)(p + 16);
  return r;
}

// gfx1250 async global->LDS copies (ASYNCcnt). LDS offset = low 32 bits of the
// generic shared-memory pointer (flat LDS aperture keeps offset in addr[31:0]).
__device__ __forceinline__ void async_g2l_b32(void* lds, const void* gptr) {
  unsigned l = (unsigned)(size_t)lds;
  unsigned long long g = (unsigned long long)(size_t)gptr;
  asm volatile("global_load_async_to_lds_b32 %0, %1, off"
               :: "v"(l), "v"(g) : "memory");
}
__device__ __forceinline__ void async_g2l_b128(void* lds, const void* gptr) {
  unsigned l = (unsigned)(size_t)lds;
  unsigned long long g = (unsigned long long)(size_t)gptr;
  asm volatile("global_load_async_to_lds_b128 %0, %1, off"
               :: "v"(l), "v"(g) : "memory");
}
__device__ __forceinline__ void wait_async0() {
  asm volatile("s_wait_asynccnt 0x0" ::: "memory");
}

// ---------------------------------------------------------------------------
// Pass 0a: split-f16 DFT matrices.
//   CWt[w'][w] =  cos(2pi w w'/64)          (fwd row-DFT, real, transposed)
//   SWt[w'][w] = -sin(2pi w w'/64)          (fwd row-DFT, imag)
//   CH [h'][h] =  cos(2pi h h'/64)          (fwd/inv col-DFT; symmetric)
//   SH [h'][h] =  sin(2pi h h'/64)
//   CIt[w][w'] =  a/4096 *  cos(2pi w w'/64), a=1 for w'=0,32 else 2; 0 if w'>=33
//   SIt[w][w'] =  a/4096 * -sin(2pi w w'/64);                         0 if w'>=33
// ---------------------------------------------------------------------------
__global__ __launch_bounds__(256) void cssm_init_mats(
    half_t* CWt_hi, half_t* CWt_lo, half_t* SWt_hi, half_t* SWt_lo,
    half_t* CH_hi,  half_t* CH_lo,  half_t* SH_hi,  half_t* SH_lo,
    half_t* CIt_hi, half_t* CIt_lo, half_t* SIt_hi, half_t* SIt_lo) {
  int i = blockIdx.x * 256 + threadIdx.x;
  const float step = 6.2831853071795864769f / 64.0f;
  if (i < WFP * 64) {                       // [w'][w], ld 64
    int w = i & 63, wp = i >> 6;
    float a = step * (float)((w * wp) & 63);
    splitf(cosf(a),  CWt_hi[i], CWt_lo[i]);
    splitf(-sinf(a), SWt_hi[i], SWt_lo[i]);
  }
  if (i < 64 * 64) {
    {                                       // CH/SH [h'][h]
      int h = i & 63, hp = i >> 6;
      float a = step * (float)((h * hp) & 63);
      splitf(cosf(a), CH_hi[i], CH_lo[i]);
      splitf(sinf(a), SH_hi[i], SH_lo[i]);
    }
    {                                       // CIt/SIt [w][w'], ld 64, pad zero
      int wp = i & 63, w = i >> 6;
      if (wp < WFD) {
        float amp = ((wp == 0) || (wp == 32)) ? 1.0f : 2.0f;
        amp *= (1.0f / 4096.0f);
        float a = step * (float)((w * wp) & 63);
        splitf(amp * cosf(a),  CIt_hi[i], CIt_lo[i]);
        splitf(-amp * sinf(a), SIt_hi[i], SIt_lo[i]);
      } else {
        CIt_hi[i] = (half_t)0.f; CIt_lo[i] = (half_t)0.f;
        SIt_hi[i] = (half_t)0.f; SIt_lo[i] = (half_t)0.f;
      }
    }
  }
}

// ---------------------------------------------------------------------------
// Pass 0b: K_hat[c,h',w'] = sum_{j,k} kernel[c,j,k] * e^{-2pi i((j+24)h'+(k+24)w')/64}
// ---------------------------------------------------------------------------
__global__ __launch_bounds__(256) void cssm_khat(const float* __restrict__ kern,
                                                 float2* __restrict__ K) {
  int idx = blockIdx.x * 256 + threadIdx.x;
  if (idx >= (int)K_ELEMS) return;
  int wp = idx % WFD;
  int r  = idx / WFD;
  int hp = r & 63;
  int c  = r >> 6;
  const float step = 6.2831853071795864769f / 64.0f;
  float sr = 0.f, si = 0.f;
  for (int j = 0; j < 15; ++j) {
    int phj = ((j + 24) * hp) & 63;
    for (int k = 0; k < 15; ++k) {
      float v = kern[(c * 15 + j) * 15 + k];
      int ph = (phj + (k + 24) * wp) & 63;
      float a = step * (float)ph;
      float sn, cs;
      sincosf(a, &sn, &cs);
      sr += v * cs;
      si -= v * sn;
    }
  }
  K[idx] = make_float2(sr, si);
}

// ---------------------------------------------------------------------------
// Pass 1: forward rfft2 per image (one workgroup = one (b,t,c) image, 8 waves).
//   Step1: Y = x * Fw   (M=h 64, N=w' 48, K=w 64), transposed+split in LDS
//   Step2: U = Fh * Y   (M=h' 64, N=w' 48, K=h 64), cols <33 -> workspace
// U layout: U[t][(b*C+c)*H + h'][w']  (bin-contiguous for the scan kernel)
// ---------------------------------------------------------------------------
__global__ __launch_bounds__(256) void cssm_fwd(
    const float* __restrict__ x,
    const half_t* __restrict__ CWt_hi, const half_t* __restrict__ CWt_lo,
    const half_t* __restrict__ SWt_hi, const half_t* __restrict__ SWt_lo,
    const half_t* __restrict__ CH_hi,  const half_t* __restrict__ CH_lo,
    const half_t* __restrict__ SH_hi,  const half_t* __restrict__ SH_lo,
    float2* __restrict__ U) {
  __shared__ __attribute__((aligned(16))) float  xstage[4096];
  __shared__ __attribute__((aligned(16))) half_t xhi[4096], xlo[4096];
  __shared__ __attribute__((aligned(16))) half_t yrh[WFP * 64], yrl[WFP * 64];
  __shared__ __attribute__((aligned(16))) half_t yih[WFP * 64], yil[WFP * 64];

  int bid = blockIdx.x;
  int c   = bid & 63;
  int bt  = bid >> 6;                 // b*64 + t
  int tid = threadIdx.x;
  const float* img = x + (size_t)bt * (64 * 64 * 64) + c;  // (h,w) at stride 64

  // Async DMA the channel-strided image into LDS (per-lane scatter-to-LDS).
  for (int i = tid; i < 4096; i += 256)
    async_g2l_b32(xstage + i, img + (size_t)i * 64);
  wait_async0();
  __syncthreads();
  for (int i = tid; i < 4096; i += 256)
    splitf(xstage[i], xhi[i], xlo[i]);
  __syncthreads();

  int wave = tid >> 5, lane = tid & 31;

  // Step 1: row DFT along w. 12 output tiles (4 x 3).
  for (int tt = wave; tt < 12; tt += 8) {
    int mr = (tt & 3) << 4;           // h tile
    int nb = (tt >> 2) << 4;          // w' tile
    v8f aR = VZERO8, aI = VZERO8;
    for (int kb = 0; kb < 64; kb += 32) {
      v16h Ah = load_a16(xhi, 64, mr, kb, lane);
      v16h Al = load_a16(xlo, 64, mr, kb, lane);
      v16h Bh = load_b16(CWt_hi, 64, kb, nb, lane);
      v16h Bl = load_b16(CWt_lo, 64, kb, nb, lane);
      aR = mma_split(Ah, Al, Bh, Bl, aR);
      Bh = load_b16(SWt_hi, 64, kb, nb, lane);
      Bl = load_b16(SWt_lo, 64, kb, nb, lane);
      aI = mma_split(Ah, Al, Bh, Bl, aI);
    }
    int colw  = nb + (lane & 15);
    int rbase = mr + ((lane >> 4) << 3);
    for (int rr = 0; rr < 8; ++rr) {   // store transposed + re-split
      int h = rbase + rr;
      splitf(aR[rr], yrh[colw * 64 + h], yrl[colw * 64 + h]);
      splitf(aI[rr], yih[colw * 64 + h], yil[colw * 64 + h]);
    }
  }
  __syncthreads();

  // Step 2: column DFT along h. Ur = CH*Yr + SH*Yi ; Ui = CH*Yi - SH*Yr
  size_t ubase = (size_t)(bt & 63) * NBINS
               + (size_t)((bt >> 6) * 64 + c) * (64 * WFD);
  for (int tt = wave; tt < 12; tt += 8) {
    int mr = (tt & 3) << 4;           // h' tile
    int nb = (tt >> 2) << 4;          // w' tile
    v8f aR = VZERO8, aI = VZERO8;
    for (int kb = 0; kb < 64; kb += 32) {
      v16h Ch = load_a16(CH_hi, 64, mr, kb, lane);
      v16h Cl = load_a16(CH_lo, 64, mr, kb, lane);
      v16h Sh = load_a16(SH_hi, 64, mr, kb, lane);
      v16h Sl = load_a16(SH_lo, 64, mr, kb, lane);
      v16h Rh = load_b16(yrh, 64, kb, nb, lane);
      v16h Rl = load_b16(yrl, 64, kb, nb, lane);
      v16h Ih = load_b16(yih, 64, kb, nb, lane);
      v16h Il = load_b16(yil, 64, kb, nb, lane);
      aR = mma_split(Ch, Cl, Rh, Rl, aR);
      aR = mma_split(Sh, Sl, Ih, Il, aR);
      aI = mma_split(Ch, Cl, Ih, Il, aI);
      aI = mma_split(-Sh, -Sl, Rh, Rl, aI);   // f16 WMMA has no A/B NEG: negate frag
    }
    int colw = nb + (lane & 15);
    if (colw < WFD) {
      int rbase = mr + ((lane >> 4) << 3);
      for (int rr = 0; rr < 8; ++rr) {
        int hp = rbase + rr;
        U[ubase + (size_t)hp * WFD + colw] = make_float2(aR[rr], aI[rr]);
      }
    }
  }
}

// ---------------------------------------------------------------------------
// Pass 2: per-bin complex recurrence over T (== reference log-domain scan).
// ---------------------------------------------------------------------------
__global__ __launch_bounds__(256) void cssm_scan(float2* __restrict__ U,
                                                 const float2* __restrict__ K) {
  size_t bin = (size_t)blockIdx.x * 256 + threadIdx.x;
  if (bin >= NBINS) return;
  float2 k = K[bin % (size_t)(CDIM * HDIM * WFD)];  // drop b
  float hr = 0.f, hi = 0.f;
  float2* p = U + bin;
  for (int t = 0; t < TDIM; ++t) {
    float2 u = p[(size_t)t * NBINS];
    float nr = k.x * hr - k.y * hi + u.x;
    float ni = k.x * hi + k.y * hr + u.y;
    hr = nr; hi = ni;
    p[(size_t)t * NBINS] = make_float2(hr, hi);
  }
}

// ---------------------------------------------------------------------------
// Pass 3: irfft2 per image.
//   StepA: Z = conj(Fh) * Xhat   (M=h, N=w' 48, K=h' 64)
//   StepB: out = Zr*CI + Zi*SI   (M=h, N=w 64, K=w' padded 64)
// The Xhat staging buffer aliases the Z arrays (Z written only after staging
// is fully consumed).
// ---------------------------------------------------------------------------
__global__ __launch_bounds__(256) void cssm_inv(
    const float2* __restrict__ U,
    const half_t* __restrict__ CH_hi,  const half_t* __restrict__ CH_lo,
    const half_t* __restrict__ SH_hi,  const half_t* __restrict__ SH_lo,
    const half_t* __restrict__ CIt_hi, const half_t* __restrict__ CIt_lo,
    const half_t* __restrict__ SIt_hi, const half_t* __restrict__ SIt_lo,
    float* __restrict__ out) {
  __shared__ __attribute__((aligned(16))) half_t xrh[WFP * 64], xrl[WFP * 64];
  __shared__ __attribute__((aligned(16))) half_t xih[WFP * 64], xil[WFP * 64];
  __shared__ __attribute__((aligned(16))) half_t zbuf[4 * 4096];   // 32 KB
  half_t* zrh = zbuf;
  half_t* zrl = zbuf + 4096;
  half_t* zih = zbuf + 8192;
  half_t* zil = zbuf + 12288;
  float2* stage = (float2*)zbuf;      // 2112 * 8B = 16.9 KB, dies before Z writes

  int bid = blockIdx.x;
  int c   = bid & 63;
  int bt  = bid >> 6;
  int tid = threadIdx.x;

  size_t ubase = (size_t)(bt & 63) * NBINS
               + (size_t)((bt >> 6) * 64 + c) * (64 * WFD);

  // Async DMA the contiguous Xhat tile (2112 float2) into LDS, 16B per lane.
  for (int j = tid; j < (64 * WFD) / 2; j += 256)
    async_g2l_b128(stage + 2 * j, U + ubase + 2 * j);
  wait_async0();
  __syncthreads();

  for (int i = tid; i < 64 * WFD; i += 256) {   // transpose + split
    int hp = i / WFD;
    int wp = i - hp * WFD;
    float2 v = stage[i];
    int idx = wp * 64 + hp;
    splitf(v.x, xrh[idx], xrl[idx]);
    splitf(v.y, xih[idx], xil[idx]);
  }
  for (int i = WFD * 64 + tid; i < WFP * 64; i += 256) {  // zero pad rows w'=33..47
    xrh[i] = (half_t)0.f; xrl[i] = (half_t)0.f;
    xih[i] = (half_t)0.f; xil[i] = (half_t)0.f;
  }
  __syncthreads();                    // staging fully consumed; Z region free

  int wave = tid >> 5, lane = tid & 31;

  for (int i = tid; i < 64 * 16; i += 256) {    // zero Z pad cols 48..63
    int idx = ((i >> 4) << 6) + 48 + (i & 15);
    zrh[idx] = (half_t)0.f; zrl[idx] = (half_t)0.f;
    zih[idx] = (half_t)0.f; zil[idx] = (half_t)0.f;
  }

  // StepA: Zr = CH*Xr - SH*Xi ; Zi = CH*Xi + SH*Xr
  for (int tt = wave; tt < 12; tt += 8) {
    int mr = (tt & 3) << 4;
    int nb = (tt >> 2) << 4;
    v8f aR = VZERO8, aI = VZERO8;
    for (int kb = 0; kb < 64; kb += 32) {
      v16h Ch = load_a16(CH_hi, 64, mr, kb, lane);
      v16h Cl = load_a16(CH_lo, 64, mr, kb, lane);
      v16h Sh = load_a16(SH_hi, 64, mr, kb, lane);
      v16h Sl = load_a16(SH_lo, 64, mr, kb, lane);
      v16h Rh = load_b16(xrh, 64, kb, nb, lane);
      v16h Rl = load_b16(xrl, 64, kb, nb, lane);
      v16h Ih = load_b16(xih, 64, kb, nb, lane);
      v16h Il = load_b16(xil, 64, kb, nb, lane);
      aR = mma_split(Ch, Cl, Rh, Rl, aR);
      aR = mma_split(-Sh, -Sl, Ih, Il, aR);
      aI = mma_split(Ch, Cl, Ih, Il, aI);
      aI = mma_split(Sh, Sl, Rh, Rl, aI);
    }
    int colw  = nb + (lane & 15);
    int rbase = mr + ((lane >> 4) << 3);
    for (int rr = 0; rr < 8; ++rr) {    // Z row-major
      int idx = (rbase + rr) * 64 + colw;
      splitf(aR[rr], zrh[idx], zrl[idx]);
      splitf(aI[rr], zih[idx], zil[idx]);
    }
  }
  __syncthreads();

  // StepB: out[h][w] = sum_w' Zr*CIt[w][w'] + Zi*SIt[w][w']
  float* oimg = out + (size_t)bt * (64 * 64 * 64) + c;
  for (int tt = wave; tt < 16; tt += 8) {
    int mr = (tt & 3) << 4;
    int nb = (tt >> 2) << 4;
    v8f acc = VZERO8;
    for (int kb = 0; kb < 64; kb += 32) {
      v16h Ah = load_a16(zrh, 64, mr, kb, lane);
      v16h Al = load_a16(zrl, 64, mr, kb, lane);
      v16h Bh = load_b16(CIt_hi, 64, kb, nb, lane);
      v16h Bl = load_b16(CIt_lo, 64, kb, nb, lane);
      acc = mma_split(Ah, Al, Bh, Bl, acc);
      Ah = load_a16(zih, 64, mr, kb, lane);
      Al = load_a16(zil, 64, mr, kb, lane);
      Bh = load_b16(SIt_hi, 64, kb, nb, lane);
      Bl = load_b16(SIt_lo, 64, kb, nb, lane);
      acc = mma_split(Ah, Al, Bh, Bl, acc);
    }
    int colw  = nb + (lane & 15);
    int rbase = mr + ((lane >> 4) << 3);
    for (int rr = 0; rr < 8; ++rr) {
      int h = rbase + rr;
      oimg[(size_t)(h * 64 + colw) * 64] = acc[rr];
    }
  }
}

// ---------------------------------------------------------------------------
extern "C" void kernel_launch(void* const* d_in, const int* in_sizes, int n_in,
                              void* d_out, int out_size, void* d_ws, size_t ws_size,
                              hipStream_t stream) {
  (void)in_sizes; (void)n_in; (void)out_size; (void)ws_size;
  const float* x    = (const float*)d_in[0];
  const float* kern = (const float*)d_in[1];
  float* out = (float*)d_out;

  char* ws = (char*)d_ws;
  float2* U    = (float2*)ws;                       // 138,412,032 B
  float2* Khat = (float2*)(ws + U_BYTES);           //   1,081,344 B
  half_t* m    = (half_t*)(ws + U_BYTES + K_BYTES); //      90,112 B of matrices
  half_t* CWt_hi = m +     0;
  half_t* CWt_lo = m +  3072;
  half_t* SWt_hi = m +  6144;
  half_t* SWt_lo = m +  9216;
  half_t* CH_hi  = m + 12288;
  half_t* CH_lo  = m + 16384;
  half_t* SH_hi  = m + 20480;
  half_t* SH_lo  = m + 24576;
  half_t* CIt_hi = m + 28672;
  half_t* CIt_lo = m + 32768;
  half_t* SIt_hi = m + 36864;
  half_t* SIt_lo = m + 40960;

  cssm_init_mats<<<16, 256, 0, stream>>>(CWt_hi, CWt_lo, SWt_hi, SWt_lo,
                                         CH_hi, CH_lo, SH_hi, SH_lo,
                                         CIt_hi, CIt_lo, SIt_hi, SIt_lo);
  cssm_khat<<<(int)((K_ELEMS + 255) / 256), 256, 0, stream>>>(kern, Khat);
  cssm_fwd<<<BDIM * TDIM * CDIM, 256, 0, stream>>>(x,
      CWt_hi, CWt_lo, SWt_hi, SWt_lo, CH_hi, CH_lo, SH_hi, SH_lo, U);
  cssm_scan<<<(int)((NBINS + 255) / 256), 256, 0, stream>>>(U, Khat);
  cssm_inv<<<BDIM * TDIM * CDIM, 256, 0, stream>>>(U,
      CH_hi, CH_lo, SH_hi, SH_lo, CIt_hi, CIt_lo, SIt_hi, SIt_lo, out);
}
